// BaseTransformer_7473243095474
// MI455X (gfx1250) — matrix operand order repaired and checked by reading, not compiled, not run on hardware
//
#include <hip/hip_runtime.h>
#include <cstdint>
#include <cstddef>

// ---------------------------------------------------------------------------
// Types for CDNA5 WMMA (gfx1250, wave32)
// ---------------------------------------------------------------------------
typedef __bf16 bf16;
typedef __attribute__((ext_vector_type(16))) __bf16 v16bf;
typedef __attribute__((ext_vector_type(8)))  __bf16 v8bf;
typedef __attribute__((ext_vector_type(8)))  float  v8f;

union AFrag { v16bf v; v8bf h[2]; };

__device__ __forceinline__ bf16 f2bf(float f) {
  unsigned u = __builtin_bit_cast(unsigned, f);
  u += 0x7FFFu + ((u >> 16) & 1u);              // round-to-nearest-even
  unsigned short s = (unsigned short)(u >> 16);
  return __builtin_bit_cast(bf16, s);
}

#define S_TOK  2048
#define NTOK   1024
#define DMODEL 768
#define NHEAD  12
#define DHEAD  64
#define FFDIM  3072
#define D3     2304

// ---------------------------------------------------------------------------
// Weight convert + transpose: W fp32 [K, Nout] -> Wt bf16 [Nout, K]
// (makes every WMMA B-fragment a contiguous 32-byte per-lane load)
// ---------------------------------------------------------------------------
__global__ void wtrans_kernel(const float* __restrict__ W, bf16* __restrict__ Wt,
                              int K, int Nout) {
  size_t idx = (size_t)blockIdx.x * blockDim.x + threadIdx.x;
  size_t total = (size_t)K * Nout;
  if (idx >= total) return;
  int n = (int)(idx % Nout);
  int k = (int)(idx / Nout);
  Wt[(size_t)n * K + k] = f2bf(W[idx]);         // coalesced read of W
}

// ---------------------------------------------------------------------------
// LayerNorm: x f32 [S,D] -> bf16 (for GEMM input) or f32 (final output)
// One 256-thread block per token row; D = 3*256.
// ---------------------------------------------------------------------------
template<bool BF16OUT>
__global__ __launch_bounds__(256) void ln_kernel(const float* __restrict__ x,
                                                 const float* __restrict__ g,
                                                 const float* __restrict__ b,
                                                 void* __restrict__ out) {
  __shared__ float red[256];
  int row = blockIdx.x;
  int t = threadIdx.x;
  const float* xr = x + (size_t)row * DMODEL;
  float v0 = xr[t], v1 = xr[t + 256], v2 = xr[t + 512];
  red[t] = v0 + v1 + v2;
  __syncthreads();
  for (int off = 128; off > 0; off >>= 1) {
    if (t < off) red[t] += red[t + off];
    __syncthreads();
  }
  float mean = red[0] * (1.0f / DMODEL);
  __syncthreads();
  float d0 = v0 - mean, d1 = v1 - mean, d2 = v2 - mean;
  red[t] = d0 * d0 + d1 * d1 + d2 * d2;
  __syncthreads();
  for (int off = 128; off > 0; off >>= 1) {
    if (t < off) red[t] += red[t + off];
    __syncthreads();
  }
  float rstd = rsqrtf(red[0] * (1.0f / DMODEL) + 1e-5f);
  float y0 = d0 * rstd * g[t]       + b[t];
  float y1 = d1 * rstd * g[t + 256] + b[t + 256];
  float y2 = d2 * rstd * g[t + 512] + b[t + 512];
  if (BF16OUT) {
    bf16* o = (bf16*)out + (size_t)row * DMODEL;
    o[t] = f2bf(y0); o[t + 256] = f2bf(y1); o[t + 512] = f2bf(y2);
  } else {
    float* o = (float*)out + (size_t)row * DMODEL;
    o[t] = y0; o[t + 256] = y1; o[t + 512] = y2;
  }
}

// ---------------------------------------------------------------------------
// bf16 GEMM via v_wmma_f32_16x16x32_bf16.
//   C[M,Nout] = A[M,K] (bf16, row-major) * Wt[Nout,K]^T (bf16, K-contiguous)
// Each wave: 64x64 output tile = 4 (M) x 4 (N) accumulators,
// 16 WMMAs per K=32 step (256 B/lane -> 32 FLOP/B from L2). 8 waves/block.
// MODE 0: store bf16; cols >= 2D scatter into vt [B,H,Dh,N]  (qkv + fused V^T)
// MODE 1: C_f32[ix] += acc + bias[col]   (residual projections, in place)
// MODE 2: store bf16( gelu(acc + bias[col]) )  (ff1)
// ---------------------------------------------------------------------------
template<int MODE>
__global__ __launch_bounds__(256) void gemm_bf16_kernel(
    const bf16* __restrict__ A, const bf16* __restrict__ Wt,
    const float* __restrict__ bias, void* __restrict__ Cv,
    bf16* __restrict__ vt, int M, int Nout, int K, int ldc) {
  int w    = blockIdx.x * 8 + (threadIdx.x >> 5);
  int lane = threadIdx.x & 31;
  int hi = lane >> 4, lo = lane & 15;
  int mtiles = M >> 6;
  int tm = w % mtiles, tn = w / mtiles;
  int m0 = tm * 64, n0 = tn * 64;
  if (n0 >= Nout) return;

  v8f acc[4][4] = {};
  // A-fragment base: lane row = lo, K base = hi*16 (contiguous, matches B)
  const bf16* abase = A + (size_t)(m0 + lo) * K + hi * 16;
  // B-fragment base: lane col = lo, K base = hi*16 (contiguous in Wt row)
  const bf16* bp = Wt + (size_t)(n0 + lo) * K + hi * 16;

  for (int k0 = 0; k0 < K; k0 += 32) {
    AFrag a[4];
#pragma unroll
    for (int m = 0; m < 4; m++) {
      const bf16* ap = abase + (size_t)(16 * m) * K + k0;
      a[m].h[0] = *(const v8bf*)(ap);
      a[m].h[1] = *(const v8bf*)(ap + 8);
    }
#pragma unroll
    for (int n = 0; n < 4; n++) {
      v16bf bfrag = *(const v16bf*)(bp + (size_t)(16 * n) * K + k0);
#pragma unroll
      for (int m = 0; m < 4; m++) {
        acc[m][n] = __builtin_amdgcn_wmma_f32_16x16x32_bf16(
            false, a[m].v, false, bfrag, (short)0, acc[m][n], false, false);
      }
    }
  }

#pragma unroll
  for (int m = 0; m < 4; m++) {
#pragma unroll
    for (int n = 0; n < 4; n++) {
#pragma unroll
      for (int r = 0; r < 8; r++) {
        int row = m0 + m * 16 + r + 8 * hi;   // C/D layout: row = r + 8*(lane>=16)
        int col = n0 + n * 16 + lo;           // col = lane & 15
        float v = acc[m][n][r];
        size_t ix = (size_t)row * ldc + col;
        if (MODE == 0) {
          bf16 bv = f2bf(v);
          if (col < 2 * DMODEL) {
            ((bf16*)Cv)[ix] = bv;             // q, k sections: [S, 3D] layout
          } else {                            // v section: fused transpose
            int c = col - 2 * DMODEL;
            int hh = c >> 6, dh = c & 63;
            int bb = row >> 10, nn = row & 1023;
            vt[((size_t)((bb * NHEAD + hh) * DHEAD + dh)) * NTOK + nn] = bv;
          }
        } else if (MODE == 1) {
          float* C = (float*)Cv;
          C[ix] = C[ix] + v + bias[col];
        } else {
          float gx = v + bias[col];
          gx = 0.5f * gx * (1.0f + erff(gx * 0.70710678118654752f));
          ((bf16*)Cv)[ix] = f2bf(gx);
        }
      }
    }
  }
}

// ---------------------------------------------------------------------------
// Flash-style attention. Grid (N/64, H, B), block = 128 (4 waves).
// Each wave: 16 query rows, streams keys in 32-wide tiles.
//   S = Q K^T (4 WMMAs), scale + rel_bias, online softmax,
//   P -> LDS -> A-fragment, O += P V (4 WMMAs).
// ---------------------------------------------------------------------------
__global__ __launch_bounds__(128) void attn_kernel(
    const bf16* __restrict__ qkv, const bf16* __restrict__ vt,
    const float* __restrict__ rel_bias, bf16* __restrict__ o, float scale) {
  __shared__ __align__(16) bf16 pbuf[4][16 * 32];
  int wave = threadIdx.x >> 5;
  int lane = threadIdx.x & 31;
  int hi = lane >> 4, lo = lane & 15;
  int q0 = blockIdx.x * 64 + wave * 16;
  int h = blockIdx.y, b = blockIdx.z;

  const bf16* qp = qkv + (size_t)(b * NTOK) * D3 + h * DHEAD;
  const bf16* kp = qp + DMODEL;
  const bf16* vp = vt + (size_t)((b * NHEAD + h) * DHEAD) * NTOK;
  const float* rb = rel_bias + (size_t)h * NTOK * NTOK + (size_t)q0 * NTOK;

  // Q fragments (Dh = 64 -> two K=32 chunks)
  AFrag aq0, aq1;
  {
    const bf16* r = qp + (size_t)(q0 + lo) * D3 + hi * 16;
    aq0.h[0] = *(const v8bf*)(r);
    aq0.h[1] = *(const v8bf*)(r + 8);
    aq1.h[0] = *(const v8bf*)(r + 32);
    aq1.h[1] = *(const v8bf*)(r + 40);
  }

  v8f oacc[4] = {};
  float mrow[8], lrow[8];
#pragma unroll
  for (int r = 0; r < 8; r++) { mrow[r] = -1e30f; lrow[r] = 0.0f; }

  bf16* pb = pbuf[wave];

  for (int j0 = 0; j0 < NTOK; j0 += 32) {
    // ---- S = Q K^T : two 16x16 tiles (keys j0..+15, j0+16..+31) ----
    v8f s[2];
#pragma unroll
    for (int jt = 0; jt < 2; jt++) {
      const bf16* kr = kp + (size_t)(j0 + 16 * jt + lo) * D3 + hi * 16;
      v16bf b0 = *(const v16bf*)(kr);        // dh 0..31 slice for this lane
      v16bf b1 = *(const v16bf*)(kr + 32);   // dh 32..63 slice
      v8f a = {};
      a = __builtin_amdgcn_wmma_f32_16x16x32_bf16(false, aq0.v, false, b0,
                                                  (short)0, a, false, false);
      a = __builtin_amdgcn_wmma_f32_16x16x32_bf16(false, aq1.v, false, b1,
                                                  (short)0, a, false, false);
      s[jt] = a;
    }

    // ---- scale + bias + online softmax (per-row over 16-lane halves) ----
    float p0v[8], p1v[8];
#pragma unroll
    for (int r = 0; r < 8; r++) {
      int row = r + 8 * hi;
      float s0 = s[0][r] * scale + rb[(size_t)row * NTOK + j0 + lo];
      float s1 = s[1][r] * scale + rb[(size_t)row * NTOK + j0 + 16 + lo];
      float mx = fmaxf(s0, s1);
      for (int msk = 1; msk < 16; msk <<= 1)
        mx = fmaxf(mx, __shfl_xor(mx, msk, 32));
      float mnew = fmaxf(mrow[r], mx);
      float corr = __expf(mrow[r] - mnew);
      mrow[r] = mnew;
      float p0 = __expf(s0 - mnew);
      float p1 = __expf(s1 - mnew);
      float ps = p0 + p1;
      for (int msk = 1; msk < 16; msk <<= 1)
        ps += __shfl_xor(ps, msk, 32);
      lrow[r] = lrow[r] * corr + ps;
      p0v[r] = p0; p1v[r] = p1;
#pragma unroll
      for (int n = 0; n < 4; n++) oacc[n][r] = oacc[n][r] * corr;
    }

    // ---- P accumulator-layout -> LDS -> A-fragment layout ----
#pragma unroll
    for (int r = 0; r < 8; r++) {
      int row = r + 8 * hi;
      pb[row * 32 + lo]      = f2bf(p0v[r]);
      pb[row * 32 + 16 + lo] = f2bf(p1v[r]);
    }
    AFrag ap;   // same-wave DS ops are in-order: RAW through LDS is safe
    ap.h[0] = *(const v8bf*)(pb + lo * 32 + hi * 16);
    ap.h[1] = *(const v8bf*)(pb + lo * 32 + hi * 16 + 8);

    // ---- O += P @ V ----
#pragma unroll
    for (int n = 0; n < 4; n++) {
      const bf16* vr = vp + (size_t)(16 * n + lo) * NTOK + j0 + hi * 16;
      v16bf bv = *(const v16bf*)(vr);
      oacc[n] = __builtin_amdgcn_wmma_f32_16x16x32_bf16(
          false, ap.v, false, bv, (short)0, oacc[n], false, false);
    }
  }

  // ---- normalize + store bf16 into o [S, D] at head slice ----
  bf16* op = o + (size_t)(b * NTOK + q0) * DMODEL + h * DHEAD;
#pragma unroll
  for (int n = 0; n < 4; n++) {
#pragma unroll
    for (int r = 0; r < 8; r++) {
      int row = r + 8 * hi;
      float val = oacc[n][r] / lrow[r];
      op[(size_t)row * DMODEL + n * 16 + lo] = f2bf(val);
    }
  }
}

// ---------------------------------------------------------------------------
// Feature output: x f32 [B,N,D] -> out [B, D, 32, 32]
// ---------------------------------------------------------------------------
__global__ void featcopy_kernel(const float* __restrict__ x, float* __restrict__ out) {
  int idx = blockIdx.x * blockDim.x + threadIdx.x;   // B*D*N
  if (idx >= 2 * DMODEL * NTOK) return;
  int p = idx & 1023;
  int rest = idx >> 10;
  int d = rest % DMODEL;
  int b = rest / DMODEL;
  out[idx] = x[((size_t)(b * NTOK + p)) * DMODEL + d];
}

// ---------------------------------------------------------------------------
// Host orchestration
// ---------------------------------------------------------------------------
extern "C" void kernel_launch(void* const* d_in, const int* in_sizes, int n_in,
                              void* d_out, int out_size, void* d_ws, size_t ws_size,
                              hipStream_t stream) {
  (void)in_sizes; (void)n_in; (void)out_size; (void)ws_size;
  const float* x_in     = (const float*)d_in[0];
  const float* rel_bias = (const float*)d_in[1];
  const float* ln1_g    = (const float*)d_in[2];
  const float* ln1_b    = (const float*)d_in[3];
  const float* qkv_w    = (const float*)d_in[4];
  const float* out_w    = (const float*)d_in[5];
  const float* out_b    = (const float*)d_in[6];
  const float* ln2_g    = (const float*)d_in[7];
  const float* ln2_b    = (const float*)d_in[8];
  const float* ff1_w    = (const float*)d_in[9];
  const float* ff1_b    = (const float*)d_in[10];
  const float* ff2_w    = (const float*)d_in[11];
  const float* ff2_b    = (const float*)d_in[12];
  const float* lnf_g    = (const float*)d_in[13];
  const float* lnf_b    = (const float*)d_in[14];

  char* ws = (char*)d_ws;
  size_t off = 0;
  auto alloc = [&](size_t bytes) -> void* {
    void* p = ws + off;
    off += (bytes + 255) & ~(size_t)255;
    return p;
  };
  float* x_cur   = (float*)alloc((size_t)S_TOK * DMODEL * 4);
  bf16*  h_bf    = (bf16*) alloc((size_t)S_TOK * DMODEL * 2);
  bf16*  qkv_bf  = (bf16*) alloc((size_t)S_TOK * D3 * 2);
  bf16*  vt_bf   = (bf16*) alloc((size_t)S_TOK * DMODEL * 2);
  bf16*  o_bf    = (bf16*) alloc((size_t)S_TOK * DMODEL * 2);
  bf16*  ff_bf   = (bf16*) alloc((size_t)S_TOK * FFDIM * 2);
  bf16*  wt_qkv  = (bf16*) alloc((size_t)DMODEL * D3 * 2);
  bf16*  wt_out  = (bf16*) alloc((size_t)DMODEL * DMODEL * 2);
  bf16*  wt_ff1  = (bf16*) alloc((size_t)DMODEL * FFDIM * 2);
  bf16*  wt_ff2  = (bf16*) alloc((size_t)FFDIM * DMODEL * 2);

  hipMemcpyAsync(x_cur, x_in, (size_t)S_TOK * DMODEL * 4,
                 hipMemcpyDeviceToDevice, stream);

  const float scale = 0.125f;   // 1/sqrt(64)

  for (int i = 0; i < 6; i++) {
    // weight convert+transpose for this layer
    wtrans_kernel<<<(DMODEL * D3 + 255) / 256, 256, 0, stream>>>(
        qkv_w + (size_t)i * DMODEL * D3, wt_qkv, DMODEL, D3);
    wtrans_kernel<<<(DMODEL * DMODEL + 255) / 256, 256, 0, stream>>>(
        out_w + (size_t)i * DMODEL * DMODEL, wt_out, DMODEL, DMODEL);
    wtrans_kernel<<<(DMODEL * FFDIM + 255) / 256, 256, 0, stream>>>(
        ff1_w + (size_t)i * DMODEL * FFDIM, wt_ff1, DMODEL, FFDIM);
    wtrans_kernel<<<(FFDIM * DMODEL + 255) / 256, 256, 0, stream>>>(
        ff2_w + (size_t)i * FFDIM * DMODEL, wt_ff2, FFDIM, DMODEL);

    // --- attention block ---
    ln_kernel<true><<<S_TOK, 256, 0, stream>>>(x_cur, ln1_g + i * DMODEL,
                                               ln1_b + i * DMODEL, h_bf);
    // qkv (+ fused V^T): [2048,768] x [768,2304]; (32 M * 36 N)/8 = 144 blocks
    gemm_bf16_kernel<0><<<144, 256, 0, stream>>>(h_bf, wt_qkv, nullptr, qkv_bf,
                                                 vt_bf, S_TOK, D3, DMODEL, D3);
    attn_kernel<<<dim3(NTOK / 64, NHEAD, 2), 128, 0, stream>>>(
        qkv_bf, vt_bf, rel_bias, o_bf, scale);
    // out-proj + residual: (32*12)/8 = 48 blocks
    gemm_bf16_kernel<1><<<48, 256, 0, stream>>>(o_bf, wt_out, out_b + i * DMODEL,
                                                x_cur, nullptr,
                                                S_TOK, DMODEL, DMODEL, DMODEL);

    // --- FFN block ---
    ln_kernel<true><<<S_TOK, 256, 0, stream>>>(x_cur, ln2_g + i * DMODEL,
                                               ln2_b + i * DMODEL, h_bf);
    // ff1 + gelu: (32*48)/8 = 192 blocks
    gemm_bf16_kernel<2><<<192, 256, 0, stream>>>(h_bf, wt_ff1, ff1_b + i * FFDIM,
                                                 ff_bf, nullptr,
                                                 S_TOK, FFDIM, DMODEL, FFDIM);
    // ff2 + residual
    gemm_bf16_kernel<1><<<48, 256, 0, stream>>>(ff_bf, wt_ff2, ff2_b + i * DMODEL,
                                                x_cur, nullptr,
                                                S_TOK, DMODEL, FFDIM, DMODEL);

    if (i == 2)
      featcopy_kernel<<<(2 * DMODEL * NTOK + 255) / 256, 256, 0, stream>>>(
          x_cur, (float*)d_out + (size_t)S_TOK * DMODEL);
    if (i == 5)
      featcopy_kernel<<<(2 * DMODEL * NTOK + 255) / 256, 256, 0, stream>>>(
          x_cur, (float*)d_out + 2 * (size_t)S_TOK * DMODEL);
  }

  // final LN -> f32 output (xf at offset 0)
  ln_kernel<false><<<S_TOK, 256, 0, stream>>>(x_cur, lnf_g, lnf_b, (float*)d_out);
}
